// BinaryProductCodeMemoryLayer_51041391346392
// MI455X (gfx1250) — compile-verified, matrix operand order checked by reading, not run on hardware
//
#include <hip/hip_runtime.h>
#include <hip/hip_bf16.h>

// ---------------------------------------------------------------------------
// Problem constants (fixed by the reference)
// ---------------------------------------------------------------------------
#define NTOK 4096      // B*S tokens
#define DMOD 1024      // d_model
#define HH   4         // heads
#define NBK  18        // buckets
#define BDIM 16        // bucket dim
#define KDIM 288       // NB*BD
#define KNN  32
#define VDIM 512
#define RNK  512

typedef __attribute__((ext_vector_type(16))) __bf16 v16bf;
typedef __attribute__((ext_vector_type(8)))  __bf16 v8bf;
typedef __attribute__((ext_vector_type(8)))  float  v8f;

// ---------------------------------------------------------------------------
// fp32 -> bf16 conversion (grid-stride)
// ---------------------------------------------------------------------------
__global__ void cvt_f32_bf16_kernel(const float* __restrict__ in,
                                    __bf16* __restrict__ out, int n) {
    int i = blockIdx.x * blockDim.x + threadIdx.x;
    int stride = gridDim.x * blockDim.x;
    for (; i < n; i += stride) out[i] = (__bf16)in[i];
}

// ---------------------------------------------------------------------------
// WMMA bf16 GEMM:  C[M,N] = A[M,K] * Bt[N,K]^T   (+ bias over N)
//   A, Bt row-major bf16.  Optional fp32 and/or bf16 outputs.
//   Block: 256 threads = 8 waves, tile 128(M) x 128(N), K-step 32.
//   Wave w: mOff = (w&3)*32, nOff = (w>>2)*64 -> 2x4 accumulators (32x64).
//   Register double-buffering: next K-tile is loaded into VGPRs right after
//   the staging barrier, overlapping HBM latency with the WMMA phase.
// Fragment layout per CDNA5 ISA (16-bit A 16x32): lane<16 holds K{0..7,16..23},
// lane>=16 holds K{8..15,24..31}; two 16B LDS segments per fragment.
// B fragment assumed mirror layout (N indexed by lane&15).
// ---------------------------------------------------------------------------
__device__ inline v16bf load_frag_row(const __bf16* row, int seg) {
    v8bf lo = *(const v8bf*)(row + seg);
    v8bf hi = *(const v8bf*)(row + 16 + seg);
    return __builtin_shufflevector(lo, hi, 0, 1, 2, 3, 4, 5, 6, 7,
                                            8, 9, 10, 11, 12, 13, 14, 15);
}

__global__ __launch_bounds__(256)
void gemm_bf16_wmma_kernel(const __bf16* __restrict__ A,
                           const __bf16* __restrict__ Bt,
                           const float* __restrict__ bias,
                           float* __restrict__ Cf,
                           __bf16* __restrict__ Cb,
                           int M, int N, int K) {
    __shared__ __align__(16) __bf16 As[128][32];
    __shared__ __align__(16) __bf16 Bs[128][32];

    const int tid  = threadIdx.x;
    const int lane = tid & 31;
    const int wave = tid >> 5;
    const int mBase = blockIdx.y * 128;
    const int nBase = blockIdx.x * 128;
    const int mOff = (wave & 3) * 32;
    const int nOff = (wave >> 2) * 64;

    v8f acc[2][4];
#pragma unroll
    for (int mi = 0; mi < 2; ++mi)
#pragma unroll
        for (int j = 0; j < 4; ++j)
#pragma unroll
            for (int e = 0; e < 8; ++e) acc[mi][j][e] = 0.0f;

    const int sr = tid >> 2;            // staging row 0..63
    const int sc = (tid & 3) * 8;       // staging k-chunk 0,8,16,24
    const int fr  = lane & 15;
    const int seg = (lane >> 4) * 8;    // 0 or 8

    const __bf16* aPtr0 = &A[(size_t)(mBase + sr) * K + sc];
    const __bf16* aPtr1 = &A[(size_t)(mBase + sr + 64) * K + sc];
    const __bf16* bPtr0 = &Bt[(size_t)(nBase + sr) * K + sc];
    const __bf16* bPtr1 = &Bt[(size_t)(nBase + sr + 64) * K + sc];

    // prologue: stage K-tile 0 into registers
    v8bf aReg0 = *(const v8bf*)aPtr0;
    v8bf aReg1 = *(const v8bf*)aPtr1;
    v8bf bReg0 = *(const v8bf*)bPtr0;
    v8bf bReg1 = *(const v8bf*)bPtr1;

    for (int k0 = 0; k0 < K; k0 += 32) {
        *(v8bf*)&As[sr][sc]      = aReg0;
        *(v8bf*)&As[sr + 64][sc] = aReg1;
        *(v8bf*)&Bs[sr][sc]      = bReg0;
        *(v8bf*)&Bs[sr + 64][sc] = bReg1;
        __syncthreads();

        // issue next tile's global loads before the WMMA phase
        if (k0 + 32 < K) {
            aReg0 = *(const v8bf*)(aPtr0 + k0 + 32);
            aReg1 = *(const v8bf*)(aPtr1 + k0 + 32);
            bReg0 = *(const v8bf*)(bPtr0 + k0 + 32);
            bReg1 = *(const v8bf*)(bPtr1 + k0 + 32);
        }

        v16bf af0 = load_frag_row(&As[mOff + fr][0], seg);
        v16bf af1 = load_frag_row(&As[mOff + 16 + fr][0], seg);
#pragma unroll
        for (int j = 0; j < 4; ++j) {
            v16bf bfrag = load_frag_row(&Bs[nOff + 16 * j + fr][0], seg);
            acc[0][j] = __builtin_amdgcn_wmma_f32_16x16x32_bf16(
                false, af0, false, bfrag, (short)0, acc[0][j], false, false);
            acc[1][j] = __builtin_amdgcn_wmma_f32_16x16x32_bf16(
                false, af1, false, bfrag, (short)0, acc[1][j], false, false);
        }
        __syncthreads();
    }

    // epilogue: C 16x16 f32 layout -> lane n = lane&15, M = r + 8*(lane>>4)
    const int rowBase = mBase + mOff + ((lane >> 4) * 8);
    const int colBase = nBase + nOff + (lane & 15);
#pragma unroll
    for (int mi = 0; mi < 2; ++mi) {
#pragma unroll
        for (int j = 0; j < 4; ++j) {
            const int col = colBase + 16 * j;
            const float bv = bias ? bias[col] : 0.0f;
#pragma unroll
            for (int r = 0; r < 8; ++r) {
                const float v = acc[mi][j][r] + bv;
                const size_t o = (size_t)(rowBase + 16 * mi + r) * N + col;
                if (Cf) Cf[o] = v;
                if (Cb) Cb[o] = (__bf16)v;
            }
        }
    }
}

// ---------------------------------------------------------------------------
// Per-(token,head) scoring + beam search over bit flips.
// One thread per row r = n*H + h (16384 rows).  Beam kept sorted ascending;
// each step merges sorted A with sorted A+delta (two-pointer, A wins ties,
// matching jax top_k index-order tie-break).  Beam arrays live in LDS,
// strided by blockDim (64) for conflict-free access.
// ---------------------------------------------------------------------------
__global__ __launch_bounds__(64)
void score_beam_kernel(const float* __restrict__ q,      // [NTOK, H*KDIM] fp32
                       const float* __restrict__ keys,   // [H,NB,2,BD] fp32
                       int* __restrict__ idxOut,         // [NTOK, H*KNN]
                       float* __restrict__ wOut) {       // [NTOK, H*KNN]
    __shared__ float ksh[HH * NBK * 2 * BDIM];           // 2304 floats
    __shared__ float penBuf[2][KNN * 64];
    __shared__ int   mskBuf[2][KNN * 64];

    const int tid = threadIdx.x;
    for (int i = tid; i < HH * NBK * 2 * BDIM; i += 64) ksh[i] = keys[i];
    __syncthreads();

    const int r = blockIdx.x * 64 + tid;                 // < NTOK*HH
    const int n = r >> 2;
    const int h = r & 3;
    const float* qr = q + (size_t)n * (HH * KDIM) + h * KDIM;
    const float* kh = ksh + h * (NBK * 2 * BDIM);

    float deltas[NBK];
    int   code = 0;
#pragma unroll
    for (int m = 0; m < NBK; ++m) {
        float a = 0.0f, b = 0.0f;
#pragma unroll
        for (int d = 0; d < BDIM; ++d) {
            const float qv = qr[m * BDIM + d];
            a = fmaf(qv, kh[m * 32 + d], a);
            b = fmaf(qv, kh[m * 32 + 16 + d], b);
        }
        deltas[m] = fabsf(a - b);
        if (b > a) code |= (1 << m);
    }

    float* pc = &penBuf[0][tid];
    float* pn = &penBuf[1][tid];
    int*   mc = &mskBuf[0][tid];
    int*   mn = &mskBuf[1][tid];
    pc[0] = 0.0f;
    mc[0] = 0;
    int cnt = 1;

    for (int t = 0; t < NBK; ++t) {
        const float dt  = deltas[t];
        const int   bit = 1 << t;
        const int   nc  = (2 * cnt < KNN) ? 2 * cnt : KNN;
        int i = 0, j = 0;
        for (int o = 0; o < nc; ++o) {
            const float pa = (i < cnt) ? pc[i * 64] : 3.4e38f;
            const float pb = (j < cnt) ? pc[j * 64] + dt : 3.4e38f;
            if (pa <= pb) {                    // A wins ties (top_k order)
                pn[o * 64] = pa;
                mn[o * 64] = mc[i * 64];
                ++i;
            } else {
                pn[o * 64] = pb;
                mn[o * 64] = mc[j * 64] ^ bit;
                ++j;
            }
        }
        float* tp = pc; pc = pn; pn = tp;
        int*   tm = mc; mc = mn; mn = tm;
        cnt = nc;
    }

    // softmax(best - pen) over 32: best cancels; pen sorted so pen[0] is min
    const float p0 = pc[0];
    float sum = 0.0f;
#pragma unroll
    for (int k = 0; k < KNN; ++k) sum += __expf(p0 - pc[k * 64]);
    const float inv = 1.0f / sum;

    const int base = r * KNN;   // == n*(H*KNN) + h*KNN
#pragma unroll
    for (int k = 0; k < KNN; ++k) {
        wOut[base + k]   = __expf(p0 - pc[k * 64]) * inv;
        idxOut[base + k] = code ^ mc[k * 64];
    }
}

// ---------------------------------------------------------------------------
// Gather + weighted sum (the bandwidth-dominant stage, ~1.07 GB of reads):
// one 256-thread block per token; float2 per lane -> each of the 128 rows is
// one fully coalesced 2KB burst.  Prefetch 8 rows ahead (global_prefetch_b8).
// Emits y in bf16 for the final WMMA GEMM.
// ---------------------------------------------------------------------------
__global__ __launch_bounds__(256)
void gather_sum_kernel(const float* __restrict__ values,  // [TK, VDIM]
                       const int* __restrict__ idx,       // [NTOK, H*KNN]
                       const float* __restrict__ w,       // [NTOK, H*KNN]
                       __bf16* __restrict__ y) {          // [NTOK, VDIM]
    __shared__ int   is[HH * KNN];
    __shared__ float wsh[HH * KNN];
    const int n   = blockIdx.x;
    const int tid = threadIdx.x;
    if (tid < HH * KNN) {
        is[tid]  = idx[n * (HH * KNN) + tid];
        wsh[tid] = w[n * (HH * KNN) + tid];
    }
    __syncthreads();

    const float2* vp = (const float2*)values;
    float a0 = 0.0f, a1 = 0.0f;
#pragma unroll 4
    for (int k = 0; k < HH * KNN; ++k) {
        if (k + 8 < HH * KNN)
            __builtin_prefetch(&vp[(size_t)is[k + 8] * (VDIM / 2) + tid], 0, 0);
        const float2 v = vp[(size_t)is[k] * (VDIM / 2) + tid];
        const float wv = wsh[k];
        a0 = fmaf(wv, v.x, a0);
        a1 = fmaf(wv, v.y, a1);
    }
    y[(size_t)n * VDIM + 2 * tid]     = (__bf16)a0;
    y[(size_t)n * VDIM + 2 * tid + 1] = (__bf16)a1;
}

// ---------------------------------------------------------------------------
// Launch
// ---------------------------------------------------------------------------
extern "C" void kernel_launch(void* const* d_in, const int* in_sizes, int n_in,
                              void* d_out, int out_size, void* d_ws, size_t ws_size,
                              hipStream_t stream) {
    const float* x      = (const float*)d_in[0];   // [2,2048,1024]
    const float* keys   = (const float*)d_in[1];   // [4,18,2,16]
    const float* values = (const float*)d_in[2];   // [262144,512]
    const float* Wd     = (const float*)d_in[3];   // [512,1024]
    const float* bd     = (const float*)d_in[4];   // [512]
    const float* Wu     = (const float*)d_in[5];   // [1152,512]
    const float* Wp     = (const float*)d_in[6];   // [1024,512]
    float* out = (float*)d_out;                    // [2,2048,1024] fp32

    char* ws = (char*)d_ws;
    size_t off = 0;
    auto alloc = [&](size_t bytes) -> void* {
        void* p = (void*)(ws + off);
        off += (bytes + 255) & ~(size_t)255;
        return p;
    };
    __bf16* xb  = (__bf16*)alloc((size_t)NTOK * DMOD * 2);          // 8 MB
    __bf16* Wdb = (__bf16*)alloc((size_t)RNK * DMOD * 2);           // 1 MB
    __bf16* Wub = (__bf16*)alloc((size_t)(HH * KDIM) * RNK * 2);    // 1.1 MB
    __bf16* Wpb = (__bf16*)alloc((size_t)DMOD * VDIM * 2);          // 1 MB
    __bf16* qdb = (__bf16*)alloc((size_t)NTOK * RNK * 2);           // 4 MB
    float*  qf  = (float*)alloc((size_t)NTOK * HH * KDIM * 4);      // 18 MB
    int*    idw = (int*)alloc((size_t)NTOK * HH * KNN * 4);         // 2 MB
    float*  wgw = (float*)alloc((size_t)NTOK * HH * KNN * 4);       // 2 MB
    __bf16* yb  = (__bf16*)alloc((size_t)NTOK * VDIM * 2);          // 4 MB

    // 1) fp32 -> bf16 conversions
    cvt_f32_bf16_kernel<<<1024, 256, 0, stream>>>(x,  xb,  NTOK * DMOD);
    cvt_f32_bf16_kernel<<<512,  256, 0, stream>>>(Wd, Wdb, RNK * DMOD);
    cvt_f32_bf16_kernel<<<512,  256, 0, stream>>>(Wu, Wub, HH * KDIM * RNK);
    cvt_f32_bf16_kernel<<<512,  256, 0, stream>>>(Wp, Wpb, DMOD * VDIM);

    // 2) qd = x @ Wd.T + bd           [4096,512]  (bf16 out for next GEMM)
    gemm_bf16_wmma_kernel<<<dim3(RNK / 128, NTOK / 128), 256, 0, stream>>>(
        xb, Wdb, bd, nullptr, qdb, NTOK, RNK, DMOD);

    // 3) q = qd @ Wu.T                [4096,1152] fp32
    gemm_bf16_wmma_kernel<<<dim3(HH * KDIM / 128, NTOK / 128), 256, 0, stream>>>(
        qdb, Wub, nullptr, qf, nullptr, NTOK, HH * KDIM, RNK);

    // 4) bucket scores + beam search + softmax -> idx, weights
    score_beam_kernel<<<(NTOK * HH) / 64, 64, 0, stream>>>(qf, keys, idw, wgw);

    // 5) weighted gather over values table -> y bf16   [4096,512]
    gather_sum_kernel<<<NTOK, 256, 0, stream>>>(values, idw, wgw, yb);

    // 6) out = y @ Wp.T               [4096,1024] fp32 -> d_out
    gemm_bf16_wmma_kernel<<<dim3(DMOD / 128, NTOK / 128), 256, 0, stream>>>(
        yb, Wpb, nullptr, out, nullptr, NTOK, DMOD, VDIM);
}
